// MultiHeadAttention_LogSparse_33758442946702
// MI455X (gfx1250) — compile-verified, matrix operand order checked by reading, not run on hardware
//
#include <hip/hip_runtime.h>
#include <hip/hip_bf16.h>

// ---------------------------------------------------------------------------
// MultiHeadAttention (log-sparse variant reference, dense eval path)
// B=8, S=1024, HID=512, H=8  (per-head dim == HID == 512)
// All matmuls via v_wmma_f32_16x16x32_f16 (f16 in, f32 accumulate).
// GEMM tiles staged with CDNA5 async-to-LDS loads (ASYNCcnt), double-buffered.
// Attention scores never touch HBM: 16x1024 f32 stripe lives in LDS.
// ---------------------------------------------------------------------------

typedef __attribute__((ext_vector_type(16))) _Float16 v16h;
typedef __attribute__((ext_vector_type(8)))  float    v8f;

#define HIDN   512
#define SEQ    1024
#define NBAT   8
#define NHEAD  8
#define QKLD   8192   // row stride (halves) of the fused q|k buffer
#define VLD    4096   // row stride of v / ctx buffers (H*HID)
#define MROWS  8192   // B*S

// ---------------------------------------------------------------------------
// WMMA helpers (CDNA5 16-bit operand layout, wave32)
//   fragment element (sel, k) lives at base + sel*sel_stride + k*k_stride
//   lanes 0..15 : sel = lane,  K in {0..7, 16..23}
//   lanes 16..31: sel = lane-16, K in {8..15, 24..31}
// ---------------------------------------------------------------------------
__device__ __forceinline__ v16h frag16(const _Float16* __restrict__ base,
                                       int sel_stride, int k_stride) {
  const int lane = threadIdx.x & 31;
  const int sel  = lane & 15;
  const int k0   = (lane >> 4) << 3;   // 0 or 8
  const _Float16* p = base + (size_t)sel * sel_stride;
  v16h f;
#pragma unroll
  for (int i = 0; i < 8; ++i) f[i]     = p[(size_t)(k0 + i)      * k_stride];
#pragma unroll
  for (int i = 0; i < 8; ++i) f[8 + i] = p[(size_t)(k0 + 16 + i) * k_stride];
  return f;
}

__device__ __forceinline__ v8f wmma16(v16h a, v16h b, v8f c) {
  return __builtin_amdgcn_wmma_f32_16x16x32_f16(false, a, false, b,
                                                (short)0, c, false, false);
}

__device__ __forceinline__ v8f zero8() {
  v8f z;
#pragma unroll
  for (int i = 0; i < 8; ++i) z[i] = 0.0f;
  return z;
}

// CDNA5 async global->LDS copy, 16B per lane (tracked by ASYNCcnt).
// ldsoff: LDS byte offset (low 32 bits of the flat pointer == LDS offset);
// gp:     per-lane global byte address.
__device__ __forceinline__ void async_ld_b128(unsigned ldsoff, const void* gp) {
  asm volatile("global_load_async_to_lds_b128 %0, %1, off"
               :: "v"(ldsoff), "v"(gp) : "memory");
}
__device__ __forceinline__ void wait_async_le4() {
  asm volatile("s_wait_asynccnt 0x4" ::: "memory");
}
__device__ __forceinline__ void wait_async_0() {
  asm volatile("s_wait_asynccnt 0x0" ::: "memory");
}
__device__ __forceinline__ unsigned lds_off(const void* p) {
  return (unsigned)(uintptr_t)p;   // flat low 32 bits == LDS byte offset
}

// ---------------------------------------------------------------------------
// f32 -> f16 conversion (grid-stride)
// ---------------------------------------------------------------------------
__global__ void f32_to_f16_kernel(const float* __restrict__ in,
                                  _Float16* __restrict__ out, size_t n) {
  size_t i = (size_t)blockIdx.x * blockDim.x + threadIdx.x;
  size_t stride = (size_t)gridDim.x * blockDim.x;
  for (; i < n; i += stride) out[i] = (_Float16)in[i];
}

// ---------------------------------------------------------------------------
// Tiled WMMA GEMM: C[M,N] = A[M,K](f16) * B[K,N](f16) + bias[N](f32)
// Block tile 128x128, K-step 32, 8 waves; wave tile 64x32 (4x2 subtiles).
// LDS tiles double-buffered, filled by async-to-LDS loads (4 per wave/stage).
// M,N multiples of 128; K multiple of 32 (true for all call sites).
// ---------------------------------------------------------------------------
#define TBM 128
#define TBN 128
#define TBK 32
#define SAS 40    // padded LDS row stride for A tile (halves, 16B aligned)
#define SBS 136   // padded LDS row stride for B tile

template <bool OUT_F32>
__global__ __launch_bounds__(256) void gemm_bias_kernel(
    const _Float16* __restrict__ A, const _Float16* __restrict__ Bm,
    const float* __restrict__ bias, void* __restrict__ C,
    int M, int N, int K) {
  __shared__ __align__(16) _Float16 sA[2][TBM * SAS];
  __shared__ __align__(16) _Float16 sB[2][TBK * SBS];

  const int tid  = threadIdx.x;
  const int wave = tid >> 5;
  const int wm   = wave & 1;      // 0..1 : 64-row block within tile
  const int wn   = wave >> 1;     // 0..3 : 32-col block within tile
  const int bm   = blockIdx.y * TBM;
  const int bn   = blockIdx.x * TBN;

  // async staging geometry: 16B chunks.
  // A tile: 128 rows x 64B  -> 512 chunks -> 2 rounds of 256 threads
  // B tile:  32 rows x 256B -> 512 chunks -> 2 rounds of 256 threads
  const int ar0 = tid >> 2,         as0 = tid & 3;
  const int ar1 = (256 + tid) >> 2, as1 = (256 + tid) & 3;
  const int br0 = tid >> 4,         bs0 = tid & 15;
  const int br1 = (256 + tid) >> 4, bs1 = (256 + tid) & 15;

  auto stage = [&](int k0, int buf) {
    async_ld_b128(lds_off(&sA[buf][ar0 * SAS + as0 * 8]),
                  &A[(size_t)(bm + ar0) * K + k0 + as0 * 8]);
    async_ld_b128(lds_off(&sA[buf][ar1 * SAS + as1 * 8]),
                  &A[(size_t)(bm + ar1) * K + k0 + as1 * 8]);
    async_ld_b128(lds_off(&sB[buf][br0 * SBS + bs0 * 8]),
                  &Bm[(size_t)(k0 + br0) * N + bn + bs0 * 8]);
    async_ld_b128(lds_off(&sB[buf][br1 * SBS + bs1 * 8]),
                  &Bm[(size_t)(k0 + br1) * N + bn + bs1 * 8]);
  };

  v8f acc[4][2];
#pragma unroll
  for (int i = 0; i < 4; ++i)
#pragma unroll
    for (int j = 0; j < 2; ++j) acc[i][j] = zero8();

  const int nk = K / TBK;
  stage(0, 0);
  for (int ks = 0; ks < nk; ++ks) {
    const int cur = ks & 1;
    if (ks + 1 < nk) {
      stage((ks + 1) * TBK, cur ^ 1);  // prefetch next tile into other buffer
      wait_async_le4();                // previous stage (ours) has landed
    } else {
      wait_async_0();
    }
    __syncthreads();                   // all waves' stage data visible

    v16h af[4], bf[2];
#pragma unroll
    for (int i = 0; i < 4; ++i)
      af[i] = frag16(&sA[cur][(wm * 64 + i * 16) * SAS], SAS, 1);
#pragma unroll
    for (int j = 0; j < 2; ++j)
      bf[j] = frag16(&sB[cur][wn * 32 + j * 16], 1, SBS);
#pragma unroll
    for (int i = 0; i < 4; ++i)
#pragma unroll
      for (int j = 0; j < 2; ++j)
        acc[i][j] = wmma16(af[i], bf[j], acc[i][j]);
    __syncthreads();                   // done reading before buffer reuse
  }

  const int lane = tid & 31;
  const int n_   = lane & 15;
  const int m0   = (lane >> 4) * 8;
#pragma unroll
  for (int i = 0; i < 4; ++i) {
#pragma unroll
    for (int j = 0; j < 2; ++j) {
      const int gn  = bn + wn * 32 + j * 16 + n_;
      const float bvv = bias ? bias[gn] : 0.0f;
      const int gm0 = bm + wm * 64 + i * 16 + m0;
#pragma unroll
      for (int e = 0; e < 8; ++e) {
        const float val = acc[i][j][e] + bvv;
        if (OUT_F32)
          reinterpret_cast<float*>(C)[(size_t)(gm0 + e) * N + gn] = val;
        else
          reinterpret_cast<_Float16*>(C)[(size_t)(gm0 + e) * N + gn] =
              (_Float16)val;
      }
    }
  }
}

// ---------------------------------------------------------------------------
// Fused attention per (batch, head, 16-row q tile):
//   pass1: scores(16x1024) = (q kT + bias) / sqrt(512)  -> LDS f32
//   pass2: exact softmax (16 lanes per row), P -> same LDS as f16
//   pass3: ctx(16x512) = P @ V, each wave owns a 64-wide d chunk
// K/V fragments are consumed straight from global: zero intra-block reuse,
// so LDS staging would only add traffic.
// ---------------------------------------------------------------------------
__global__ __launch_bounds__(256) void attn_kernel(
    const _Float16* __restrict__ qkh, const _Float16* __restrict__ vmat,
    const float* __restrict__ bias, _Float16* __restrict__ ctxh) {
  __shared__ __align__(16) float sS[16 * SEQ];  // 64 KB, reused as f16 P

  const int q0   = blockIdx.x * 16;
  const int h    = blockIdx.y;
  const int bz   = blockIdx.z;
  const int tid  = threadIdx.x;
  const int w    = tid >> 5;        // wave 0..7
  const int lane = tid & 31;
  const int n_   = lane & 15;
  const int m0   = (lane >> 4) * 8;

  const _Float16* qbase = qkh + (size_t)(bz * SEQ + q0) * QKLD + h * HIDN;
  const _Float16* kbase = qkh + (size_t)(bz * SEQ) * QKLD + NHEAD * HIDN + h * HIDN;

  // ---- pass 1: wave w computes kv columns [w*128, w*128+128) ----
  {
    v8f acc[8];
#pragma unroll
    for (int t = 0; t < 8; ++t) acc[t] = zero8();

    for (int d0 = 0; d0 < HIDN; d0 += 32) {
      v16h aq = frag16(qbase + d0, QKLD, 1);
#pragma unroll
      for (int t = 0; t < 8; ++t) {
        const int kv0 = w * 128 + t * 16;
        v16h bk = frag16(kbase + (size_t)kv0 * QKLD + d0, QKLD, 1);
        acc[t] = wmma16(aq, bk, acc[t]);
      }
    }
    const float SCALE = 0.04419417382415922f;  // 1/sqrt(512)
#pragma unroll
    for (int t = 0; t < 8; ++t) {
      const int kvc = w * 128 + t * 16 + n_;
#pragma unroll
      for (int e = 0; e < 8; ++e) {
        const int m = m0 + e;
        sS[m * SEQ + kvc] =
            (acc[t][e] + bias[(size_t)(q0 + m) * SEQ + kvc]) * SCALE;
      }
    }
  }
  __syncthreads();

  // ---- pass 2: exact softmax, 16 lanes per row (lanes stay in one wave) ----
  {
    const int row = tid >> 4;   // 0..15
    const int l16 = tid & 15;
    float vr[64];
    const float* srow = &sS[row * SEQ + l16 * 64];
#pragma unroll
    for (int i = 0; i < 64; ++i) vr[i] = srow[i];
    float mx = -3.0e38f;
#pragma unroll
    for (int i = 0; i < 64; ++i) mx = fmaxf(mx, vr[i]);
#pragma unroll
    for (int off = 8; off >= 1; off >>= 1)
      mx = fmaxf(mx, __shfl_xor(mx, off, 16));
    float sum = 0.0f;
#pragma unroll
    for (int i = 0; i < 64; ++i) { vr[i] = __expf(vr[i] - mx); sum += vr[i]; }
#pragma unroll
    for (int off = 8; off >= 1; off >>= 1) sum += __shfl_xor(sum, off, 16);
    const float inv = 1.0f / sum;
    __syncthreads();  // all reads of f32 scores complete before overwrite
    _Float16* sP = reinterpret_cast<_Float16*>(sS);
    _Float16* prow = &sP[row * SEQ + l16 * 64];
#pragma unroll
    for (int i = 0; i < 64; ++i) prow[i] = (_Float16)(vr[i] * inv);
  }
  __syncthreads();

  // ---- pass 3: P @ V, wave w owns d chunk [w*64, w*64+64) ----
  {
    const _Float16* sP = reinterpret_cast<const _Float16*>(sS);
    const _Float16* vbase =
        vmat + (size_t)(bz * SEQ) * VLD + h * HIDN + w * 64;
    v8f o[4];
#pragma unroll
    for (int s = 0; s < 4; ++s) o[s] = zero8();

    for (int kv0 = 0; kv0 < SEQ; kv0 += 32) {
      v16h ap = frag16(sP + kv0, SEQ, 1);
#pragma unroll
      for (int s = 0; s < 4; ++s) {
        v16h bvf = frag16(vbase + (size_t)kv0 * VLD + s * 16, 1, VLD);
        o[s] = wmma16(ap, bvf, o[s]);
      }
    }
#pragma unroll
    for (int s = 0; s < 4; ++s) {
      const int dcol = w * 64 + s * 16 + n_;
#pragma unroll
      for (int e = 0; e < 8; ++e) {
        const int m = m0 + e;
        ctxh[(size_t)(bz * SEQ + q0 + m) * VLD + h * HIDN + dcol] =
            (_Float16)o[s][e];
      }
    }
  }
}

// ---------------------------------------------------------------------------
// Host launch
// ---------------------------------------------------------------------------
extern "C" void kernel_launch(void* const* d_in, const int* in_sizes, int n_in,
                              void* d_out, int out_size, void* d_ws,
                              size_t ws_size, hipStream_t stream) {
  (void)in_sizes; (void)n_in; (void)out_size; (void)ws_size;
  const float* x     = (const float*)d_in[0];
  const float* abias = (const float*)d_in[1];
  const float* Wq    = (const float*)d_in[2];
  const float* bq    = (const float*)d_in[3];
  const float* Wqk   = (const float*)d_in[4];
  const float* bqk   = (const float*)d_in[5];
  const float* Wv    = (const float*)d_in[6];
  const float* bv    = (const float*)d_in[7];
  const float* Wproj = (const float*)d_in[8];
  const float* bproj = (const float*)d_in[9];
  const float* Wout  = (const float*)d_in[10];
  const float* bout  = (const float*)d_in[11];
  float* out = (float*)d_out;

  char* ws = (char*)d_ws;
  auto carve = [&](size_t nhalf) {
    _Float16* p = (_Float16*)ws;
    ws += ((nhalf * sizeof(_Float16) + 255) / 256) * 256;
    return p;
  };
  _Float16* xh     = carve((size_t)MROWS * HIDN);
  _Float16* Wqh    = carve((size_t)HIDN * HIDN);
  _Float16* Wqkh   = carve((size_t)HIDN * QKLD);
  _Float16* Wvh    = carve((size_t)HIDN * VLD);
  _Float16* Wprojh = carve((size_t)VLD * HIDN);
  _Float16* Wouth  = carve((size_t)HIDN * HIDN);
  _Float16* x1h    = carve((size_t)MROWS * HIDN);
  _Float16* qkh    = carve((size_t)MROWS * QKLD);
  _Float16* vh     = carve((size_t)MROWS * VLD);
  _Float16* ctxh   = carve((size_t)MROWS * VLD);
  _Float16* tmph   = carve((size_t)MROWS * HIDN);

  auto cvt = [&](const float* src, _Float16* dst, size_t n) {
    int blocks = (int)((n + 1023) / 1024);
    f32_to_f16_kernel<<<blocks, 256, 0, stream>>>(src, dst, n);
  };
  cvt(x, xh, (size_t)MROWS * HIDN);
  cvt(Wq, Wqh, (size_t)HIDN * HIDN);
  cvt(Wqk, Wqkh, (size_t)HIDN * QKLD);
  cvt(Wv, Wvh, (size_t)HIDN * VLD);
  cvt(Wproj, Wprojh, (size_t)VLD * HIDN);
  cvt(Wout, Wouth, (size_t)HIDN * HIDN);

  // x1 = x @ Wq_lin + bq
  gemm_bias_kernel<false><<<dim3(HIDN / TBN, MROWS / TBM), 256, 0, stream>>>(
      xh, Wqh, bq, x1h, MROWS, HIDN, HIDN);
  // qk = x1 @ W_qk + b_qk   (q | k fused along N)
  gemm_bias_kernel<false><<<dim3(QKLD / TBN, MROWS / TBM), 256, 0, stream>>>(
      x1h, Wqkh, bqk, qkh, MROWS, QKLD, HIDN);
  // v = x1 @ W_v + b_v
  gemm_bias_kernel<false><<<dim3(VLD / TBN, MROWS / TBM), 256, 0, stream>>>(
      x1h, Wvh, bv, vh, MROWS, VLD, HIDN);
  // fused softmax attention -> ctx
  attn_kernel<<<dim3(SEQ / 16, NHEAD, NBAT), 256, 0, stream>>>(
      qkh, vh, abias, ctxh);
  // tmp = ctx @ W_proj + b_proj
  gemm_bias_kernel<false><<<dim3(HIDN / TBN, MROWS / TBM), 256, 0, stream>>>(
      ctxh, Wprojh, bproj, tmph, MROWS, HIDN, VLD);
  // out = tmp @ W_out + b_out   (f32 output)
  gemm_bias_kernel<true><<<dim3(HIDN / TBN, MROWS / TBM), 256, 0, stream>>>(
      tmph, Wouth, bout, out, MROWS, HIDN, HIDN);
}
